// Net_16045997818449
// MI455X (gfx1250) — matrix operand order repaired and checked by reading, not compiled, not run on hardware
//
#include <hip/hip_runtime.h>
#include <math.h>

typedef __attribute__((ext_vector_type(16))) _Float16 v16h;
typedef __attribute__((ext_vector_type(8)))  _Float16 v8h;
typedef __attribute__((ext_vector_type(4)))  _Float16 v4h;
typedef __attribute__((ext_vector_type(8)))  float    v8f;

#define NEG_SLOPE 0.2f

// ---------------------------------------------------------------- utilities
__device__ __forceinline__ float atomic_max_float(float* addr, float val) {
  if (val >= 0.0f)
    return __int_as_float(atomicMax((int*)addr, __float_as_int(val)));
  else
    return __uint_as_float(atomicMin((unsigned int*)addr, __float_as_uint(val)));
}

__device__ __forceinline__ void edge_sd(const long long* __restrict__ ei,
                                        long long E, long long e, int& s, int& d) {
  if (e < E) { s = (int)ei[e]; d = (int)ei[E + e]; }
  else       { s = d = (int)(e - E); }   // appended self-loops
}

// 16-half fragment from two contiguous 16-byte LDS runs (-> 2x ds_load_b128)
__device__ __forceinline__ v16h load_frag16(const _Float16* p0, const _Float16* p1) {
  v8h lo = *(const v8h*)p0;
  v8h hi = *(const v8h*)p1;
  return __builtin_shufflevector(lo, hi, 0, 1, 2, 3, 4, 5, 6, 7,
                                          8, 9, 10, 11, 12, 13, 14, 15);
}

__device__ __forceinline__ v4h cvt4(float4 v) {
  return (v4h){(_Float16)v.x, (_Float16)v.y, (_Float16)v.z, (_Float16)v.w};
}

__global__ void fill_f32_k(float* __restrict__ p, float v, long long n) {
  long long i = (long long)blockIdx.x * blockDim.x + threadIdx.x;
  if (i < n) p[i] = v;
}

// ---------------------------------------------------------------- GEMM 1 (WMMA)
// X:[Nn,512] f32 -> H:[Nn,64] f32 via f16 WMMA 16x16x32. 128 rows / block.
// Software-pipelined: tile k+32 is prefetched into registers during WMMA of tile k.
__global__ __launch_bounds__(256) void gemm1_wmma_k(const float* __restrict__ X,
                                                    const float* __restrict__ W,
                                                    float* __restrict__ Hout,
                                                    int Nn) {
  __shared__ _Float16 As[128][40];   // 80B row stride: 8/16B aligned, conflict-free
  __shared__ _Float16 Bt[64][40];    // Bt[n][k] = W[k0+k][n]
  const int tid  = threadIdx.x;
  const int wave = tid >> 5;
  const int lane = tid & 31;
  const int m    = lane & 15;
  const int hi   = lane >> 4;
  const int rowBase = blockIdx.x * 128;
  const bool fullTile = (rowBase + 128 <= Nn);
  const float4* Xv = (const float4*)X;            // row stride 128 float4
  const float4* Wv = (const float4*)W;            // row stride 16 float4

  // per-thread staging coordinates (loop-invariant)
  int ar[4], ac4[4];
  #pragma unroll
  for (int j = 0; j < 4; ++j) { int i = tid + j * 256; ar[j] = i >> 3; ac4[j] = i & 7; }
  int br[2], bc4[2];
  #pragma unroll
  for (int j = 0; j < 2; ++j) { int i = tid + j * 256; br[j] = i >> 4; bc4[j] = i & 15; }

  auto loadTiles = [&](int k0, float4* xa, float4* wb) {
    const int kb = k0 >> 2;
    #pragma unroll
    for (int j = 0; j < 4; ++j) {
      xa[j] = make_float4(0.f, 0.f, 0.f, 0.f);
      if (fullTile || rowBase + ar[j] < Nn)
        xa[j] = Xv[(size_t)(rowBase + ar[j]) * 128 + kb + ac4[j]];
    }
    #pragma unroll
    for (int j = 0; j < 2; ++j)
      wb[j] = Wv[(size_t)(k0 + br[j]) * 16 + bc4[j]];
  };
  auto storeTiles = [&](const float4* xa, const float4* wb) {
    #pragma unroll
    for (int j = 0; j < 4; ++j)
      *(v4h*)&As[ar[j]][ac4[j] * 4] = cvt4(xa[j]);
    #pragma unroll
    for (int j = 0; j < 2; ++j) {     // transposed scatter (4x ds_store_b16)
      Bt[bc4[j] * 4 + 0][br[j]] = (_Float16)wb[j].x;
      Bt[bc4[j] * 4 + 1][br[j]] = (_Float16)wb[j].y;
      Bt[bc4[j] * 4 + 2][br[j]] = (_Float16)wb[j].z;
      Bt[bc4[j] * 4 + 3][br[j]] = (_Float16)wb[j].w;
    }
  };

  v8f acc[4];
  #pragma unroll
  for (int t = 0; t < 4; ++t) acc[t] = (v8f){0.f,0.f,0.f,0.f,0.f,0.f,0.f,0.f};

  float4 xa[4], wb[2];
  loadTiles(0, xa, wb);
  storeTiles(xa, wb);

  for (int k0 = 0; k0 < 512; k0 += 32) {
    __syncthreads();                                  // LDS tile k0 ready
    const bool more = (k0 + 32 < 512);
    if (more) loadTiles(k0 + 32, xa, wb);             // prefetch next tile (in flight)

    // A fragment (ISA 16-bit A 16x32): lanes<16 -> K 0-7,16-23 ; lanes>=16 -> K 8-15,24-31
    v16h a = load_frag16(&As[wave * 16 + m][8 * hi],
                         &As[wave * 16 + m][16 + 8 * hi]);
    #pragma unroll
    for (int t = 0; t < 4; ++t) {
      // B fragment: lanes<16 hold K 0-7,16-23 ; lanes>=16 hold K 8-15,24-31; N = lane&15
      v16h b = load_frag16(&Bt[t * 16 + m][8 * hi],
                           &Bt[t * 16 + m][16 + 8 * hi]);
      acc[t] = __builtin_amdgcn_wmma_f32_16x16x32_f16(false, a, false, b,
                                                      (short)0, acc[t], false, false);
    }
    __syncthreads();                                  // all reads of tile k0 done
    if (more) storeTiles(xa, wb);
  }

  #pragma unroll
  for (int t = 0; t < 4; ++t)
    #pragma unroll
    for (int i = 0; i < 8; ++i) {
      int r = rowBase + wave * 16 + i + 8 * hi;   // C/D layout: VGPR i -> M=i (+8 for hi lanes)
      if (r < Nn) Hout[(size_t)r * 64 + t * 16 + m] = acc[t][i];
    }
}

// ---------------------------------------------------------------- GEMM 2 (WMMA)
// X:[Nn,64] -> H:[Nn,16], K=64 => two 16x16x32 WMMA steps. 128 rows / block.
__global__ __launch_bounds__(256) void gemm2_wmma_k(const float* __restrict__ X,
                                                    const float* __restrict__ W,
                                                    float* __restrict__ Hout,
                                                    int Nn) {
  __shared__ _Float16 As[128][72];   // 144B row stride, aligned + conflict-free
  __shared__ _Float16 Bt[16][72];    // Bt[n][k] = W[k][n]
  const int tid  = threadIdx.x;
  const int wave = tid >> 5;
  const int lane = tid & 31;
  const int m    = lane & 15;
  const int hi   = lane >> 4;
  const int rowBase = blockIdx.x * 128;
  const bool fullTile = (rowBase + 128 <= Nn);
  const float4* Xv = (const float4*)X;            // row stride 16 float4
  const float4* Wv = (const float4*)W;            // row stride 4 float4

  // batch loads into registers first, then LDS stores
  float4 xa[8];
  #pragma unroll
  for (int j = 0; j < 8; ++j) {
    int i = tid + j * 256;
    int r = i >> 4, c4 = i & 15;
    xa[j] = make_float4(0.f, 0.f, 0.f, 0.f);
    if (fullTile || rowBase + r < Nn) xa[j] = Xv[(size_t)(rowBase + r) * 16 + c4];
  }
  float4 wb = Wv[tid & 255];                      // 64x16 = 256 float4
  #pragma unroll
  for (int j = 0; j < 8; ++j) {
    int i = tid + j * 256;
    int r = i >> 4, c4 = i & 15;
    *(v4h*)&As[r][c4 * 4] = cvt4(xa[j]);
  }
  {
    int r = tid >> 2, c4 = tid & 3;               // r=K 0..63, c4=N/4
    Bt[c4 * 4 + 0][r] = (_Float16)wb.x;
    Bt[c4 * 4 + 1][r] = (_Float16)wb.y;
    Bt[c4 * 4 + 2][r] = (_Float16)wb.z;
    Bt[c4 * 4 + 3][r] = (_Float16)wb.w;
  }
  __syncthreads();

  v8f acc = (v8f){0.f,0.f,0.f,0.f,0.f,0.f,0.f,0.f};
  #pragma unroll
  for (int ks = 0; ks < 2; ++ks) {
    v16h a = load_frag16(&As[wave * 16 + m][ks * 32 + 8 * hi],
                         &As[wave * 16 + m][ks * 32 + 16 + 8 * hi]);
    v16h b = load_frag16(&Bt[m][ks * 32 + 8 * hi],
                         &Bt[m][ks * 32 + 16 + 8 * hi]);
    acc = __builtin_amdgcn_wmma_f32_16x16x32_f16(false, a, false, b,
                                                 (short)0, acc, false, false);
  }
  #pragma unroll
  for (int i = 0; i < 8; ++i) {
    int r = rowBase + wave * 16 + i + 8 * hi;
    if (r < Nn) Hout[(size_t)r * 16 + m] = acc[i];
  }
}

// ---------------------------------------------------------------- attention coefficients
template<int H, int C>
__global__ void calc_alpha_k(const float* __restrict__ feat,
                             const float* __restrict__ asrc,
                             const float* __restrict__ adst,
                             float* __restrict__ as, float* __restrict__ ad,
                             long long Nn) {
  long long idx = (long long)blockIdx.x * blockDim.x + threadIdx.x;
  if (idx >= Nn * H) return;
  long long n = idx / H;
  int h = (int)(idx - n * H);
  float s = 0.f, d = 0.f;
  #pragma unroll
  for (int c = 0; c < C; ++c) {
    float v = feat[n * (H * C) + h * C + c];
    s += v * asrc[h * C + c];
    d += v * adst[h * C + c];
  }
  as[n * H + h] = s;
  ad[n * H + h] = d;
}

// ---------------------------------------------------------------- edge passes
template<int H>
__global__ void edge_max_k(const long long* __restrict__ ei, long long E, long long NE,
                           const float* __restrict__ as, const float* __restrict__ ad,
                           float* __restrict__ m) {
  long long idx = (long long)blockIdx.x * blockDim.x + threadIdx.x;
  if (idx >= NE * H) return;
  long long e = idx / H;
  int h = (int)(idx - e * H);
  int s, d; edge_sd(ei, E, e, s, d);
  float v = as[(long long)s * H + h] + ad[(long long)d * H + h];
  if (v < 0.f) v *= NEG_SLOPE;
  atomic_max_float(&m[(long long)d * H + h], v);
}

template<int H>
__global__ void edge_den_k(const long long* __restrict__ ei, long long E, long long NE,
                           const float* __restrict__ as, const float* __restrict__ ad,
                           const float* __restrict__ m, float* __restrict__ den) {
  long long idx = (long long)blockIdx.x * blockDim.x + threadIdx.x;
  if (idx >= NE * H) return;
  long long e = idx / H;
  int h = (int)(idx - e * H);
  int s, d; edge_sd(ei, E, e, s, d);
  float v = as[(long long)s * H + h] + ad[(long long)d * H + h];
  if (v < 0.f) v *= NEG_SLOPE;
  atomicAdd(&den[(long long)d * H + h], __expf(v - m[(long long)d * H + h]));
}

template<int H, int C>
__global__ void edge_agg_k(const long long* __restrict__ ei, long long E, long long NE,
                           const float* __restrict__ as, const float* __restrict__ ad,
                           const float* __restrict__ m, const float* __restrict__ den,
                           const float* __restrict__ feat, float* __restrict__ out) {
  long long idx = (long long)blockIdx.x * blockDim.x + threadIdx.x;
  if (idx >= NE * H) return;
  long long e = idx / H;
  int h = (int)(idx - e * H);
  int s, d; edge_sd(ei, E, e, s, d);
  float v = as[(long long)s * H + h] + ad[(long long)d * H + h];
  if (v < 0.f) v *= NEG_SLOPE;
  float w = __expf(v - m[(long long)d * H + h]) /
            (den[(long long)d * H + h] + 1e-16f);
  #pragma unroll
  for (int c = 0; c < C; ++c)
    atomicAdd(&out[(long long)d * (H * C) + h * C + c],
              feat[(long long)s * (H * C) + h * C + c] * w);
}

// ---------------------------------------------------------------- elementwise tails
__global__ void elu_bias_k(const float* __restrict__ agg, const float* __restrict__ b,
                           float* __restrict__ out, long long total, int F) {
  long long i = (long long)blockIdx.x * blockDim.x + threadIdx.x;
  if (i >= total) return;
  float v = agg[i] + b[(int)(i % F)];
  out[i] = (v > 0.f) ? v : (__expf(v) - 1.0f);
}

__global__ void bias_logsoftmax_k(const float* __restrict__ agg, const float* __restrict__ b,
                                  float* __restrict__ out, long long Nn) {
  long long n = (long long)blockIdx.x * blockDim.x + threadIdx.x;
  if (n >= Nn) return;
  float v[16];
  float mx = -3.0e38f;
  #pragma unroll
  for (int c = 0; c < 16; ++c) { v[c] = agg[n * 16 + c] + b[c]; mx = fmaxf(mx, v[c]); }
  float s = 0.f;
  #pragma unroll
  for (int c = 0; c < 16; ++c) s += __expf(v[c] - mx);
  float ls = __logf(s);
  #pragma unroll
  for (int c = 0; c < 16; ++c) out[n * 16 + c] = v[c] - mx - ls;
}

// ---------------------------------------------------------------- launcher
extern "C" void kernel_launch(void* const* d_in, const int* in_sizes, int n_in,
                              void* d_out, int out_size, void* d_ws, size_t ws_size,
                              hipStream_t stream) {
  const float*     x   = (const float*)d_in[0];
  const long long* ei  = (const long long*)d_in[1];
  const float*     W1  = (const float*)d_in[2];
  const float*     a1s = (const float*)d_in[3];
  const float*     a1d = (const float*)d_in[4];
  const float*     b1  = (const float*)d_in[5];
  const float*     W2  = (const float*)d_in[6];
  const float*     a2s = (const float*)d_in[7];
  const float*     a2d = (const float*)d_in[8];
  const float*     b2  = (const float*)d_in[9];
  float* out = (float*)d_out;

  const long long N  = in_sizes[0] / 512;
  const long long E  = in_sizes[1] / 2;
  const long long NE = E + N;

  float* ws   = (float*)d_ws;
  float* h1   = ws;   ws += N * 64;   // layer1 features (then reused for elu output)
  float* as1  = ws;   ws += N * 8;
  float* ad1  = ws;   ws += N * 8;
  float* m1   = ws;   ws += N * 8;
  float* den1 = ws;   ws += N * 8;
  float* agg1 = ws;   ws += N * 64;   // layer1 aggregate (then reused for agg2)
  float* h2f  = ws;   ws += N * 16;
  float* as2  = ws;   ws += N;
  float* ad2  = ws;   ws += N;
  float* m2   = ws;   ws += N;
  float* den2 = ws;   ws += N;
  float* agg2 = agg1;                 // reuse (agg1 dead after elu)

  auto blk = [](long long n) { return (unsigned)((n + 255) / 256); };
  const float NEG_INF = -__builtin_huge_valf();

  // ---- layer 1
  gemm1_wmma_k<<<(unsigned)((N + 127) / 128), 256, 0, stream>>>(x, W1, h1, (int)N);
  calc_alpha_k<8, 8><<<blk(N * 8), 256, 0, stream>>>(h1, a1s, a1d, as1, ad1, N);
  fill_f32_k<<<blk(N * 8), 256, 0, stream>>>(m1, NEG_INF, N * 8);
  fill_f32_k<<<blk(N * 8), 256, 0, stream>>>(den1, 0.f, N * 8);
  fill_f32_k<<<blk(N * 64), 256, 0, stream>>>(agg1, 0.f, N * 64);
  edge_max_k<8><<<blk(NE * 8), 256, 0, stream>>>(ei, E, NE, as1, ad1, m1);
  edge_den_k<8><<<blk(NE * 8), 256, 0, stream>>>(ei, E, NE, as1, ad1, m1, den1);
  edge_agg_k<8, 8><<<blk(NE * 8), 256, 0, stream>>>(ei, E, NE, as1, ad1, m1, den1, h1, agg1);
  elu_bias_k<<<blk(N * 64), 256, 0, stream>>>(agg1, b1, h1, N * 64, 64);

  // ---- layer 2
  gemm2_wmma_k<<<(unsigned)((N + 127) / 128), 256, 0, stream>>>(h1, W2, h2f, (int)N);
  calc_alpha_k<1, 16><<<blk(N), 256, 0, stream>>>(h2f, a2s, a2d, as2, ad2, N);
  fill_f32_k<<<blk(N), 256, 0, stream>>>(m2, NEG_INF, N);
  fill_f32_k<<<blk(N), 256, 0, stream>>>(den2, 0.f, N);
  fill_f32_k<<<blk(N * 16), 256, 0, stream>>>(agg2, 0.f, N * 16);
  edge_max_k<1><<<blk(NE), 256, 0, stream>>>(ei, E, NE, as2, ad2, m2);
  edge_den_k<1><<<blk(NE), 256, 0, stream>>>(ei, E, NE, as2, ad2, m2, den2);
  edge_agg_k<1, 16><<<blk(NE), 256, 0, stream>>>(ei, E, NE, as2, ad2, m2, den2, h2f, agg2);

  bias_logsoftmax_k<<<blk(N), 256, 0, stream>>>(agg2, b2, out, N);
}
